// AttentionHead_91018946937291
// MI455X (gfx1250) — matrix operand order repaired and checked by reading, not compiled, not run on hardware
//
#include <hip/hip_runtime.h>

// ---------------------------------------------------------------------------
// Causal single-head attention for MI455X (gfx1250, wave32, WMMA bf16).
//   Kernel 1: QKV projection  (x @ W{q,k,v} + b)  fp32 -> bf16, WMMA 16x16x32
//   Kernel 2: flash attention, double-buffered K/V staging with CDNA5
//             GLOBAL_LOAD_ASYNC_TO_LDS (ASYNCcnt) when available.
// ---------------------------------------------------------------------------

#define MODEL_DIM 1024
#define HEAD_DIM  64
#define BATCH     4
#define SEQ       4096

typedef __attribute__((ext_vector_type(16))) __bf16 v16bf;
typedef __attribute__((ext_vector_type(8)))  float  v8f;
typedef __attribute__((ext_vector_type(4)))  int    v4i;

union BfFrag {
  v16bf    v;
  unsigned u[8];
  __bf16   h[16];
};

__device__ __forceinline__ __bf16 f2bf(float f) { return (__bf16)f; }

// CDNA5 async global->LDS copy (tracked by ASYNCcnt). The clang builtin takes
// typed v4i pointers in addrspace(1)/addrspace(3). Guarded so the kernel
// still compiles (with a plain b128 copy) if the toolchain lacks the builtin.
#if __has_builtin(__builtin_amdgcn_global_load_async_to_lds_b128)
#define HAVE_ASYNC_LDS 1
typedef __attribute__((address_space(1))) v4i gbl_v4i;
typedef __attribute__((address_space(3))) v4i lds_v4i;
// Generic LDS addresses carry the shared aperture in the high bits; the low
// 32 bits are the LDS offset (ISA 10.2), so an integer round-trip yields a
// valid as3 pointer. Generic global addresses equal as1 addresses.
__device__ __forceinline__ lds_v4i* to_lds(void* p) {
  return (lds_v4i*)(unsigned)(unsigned long long)p;
}
__device__ __forceinline__ gbl_v4i* to_gbl(const void* p) {
  return (gbl_v4i*)(unsigned long long)p;
}
#else
#define HAVE_ASYNC_LDS 0
#endif

__device__ __forceinline__ void wait_async_zero() {
#if HAVE_ASYNC_LDS
#if __has_builtin(__builtin_amdgcn_s_wait_asynccnt)
  __builtin_amdgcn_s_wait_asynccnt(0);
#else
  asm volatile("s_wait_asynccnt 0" ::: "memory");
#endif
#endif
}

// ---------------------------------------------------------------------------
// Kernel 1: QKV projection. Block = 256 threads (8 waves). Each wave computes
// a 16-row x 64-col tile of Q, K and V (x is read exactly once from HBM).
// Weight K-chunks are staged transposed in LDS as bf16 so WMMA B-fragments
// are contiguous 32B LDS loads.
// ---------------------------------------------------------------------------
__global__ __launch_bounds__(256) void qkv_proj_kernel(
    const float* __restrict__ x,
    const float* __restrict__ Wk, const float* __restrict__ bk,
    const float* __restrict__ Wq, const float* __restrict__ bq,
    const float* __restrict__ Wv, const float* __restrict__ bv,
    __bf16* __restrict__ Qo, __bf16* __restrict__ Ko, __bf16* __restrict__ Vo)
{
  __shared__ __align__(32) __bf16 Wt[3][HEAD_DIM][32];  // [mat][n][k] 12 KB

  const int  tid  = threadIdx.x;
  const int  lane = tid & 31;
  const int  wave = tid >> 5;          // 0..7
  const int  m    = lane & 15;         // A-row / B-col within tile
  const int  half = lane >> 4;         // lane half per ISA layout
  const long rowBase = (long)blockIdx.x * 128 + (long)wave * 16;

  const float* Ws[3] = {Wq, Wk, Wv};

  v8f acc[3][4] = {};                  // [matrix][n-tile] fp32 accumulators

  const float* xr = x + (rowBase + m) * (long)MODEL_DIM;

  for (int kc = 0; kc < MODEL_DIM; kc += 32) {
    __syncthreads();                   // protect previous chunk's LDS reads
    for (int w = 0; w < 3; ++w) {
      const float* W = Ws[w];
      for (int idx = tid; idx < 32 * HEAD_DIM; idx += 256) {
        int kl = idx >> 6;             // 0..31
        int n  = idx & 63;             // 0..63
        Wt[w][n][kl] = f2bf(W[(kc + kl) * HEAD_DIM + n]);
      }
    }
    __syncthreads();

    // A fragment: 16x32 bf16, contiguous order (element s: K=16h+s).
    BfFrag a;
#pragma unroll
    for (int v = 0; v < 4; ++v) {
      int k0 = kc + 16 * half + 2 * v;
      a.h[2 * v]     = f2bf(xr[k0]);
      a.h[2 * v + 1] = f2bf(xr[k0 + 1]);
      int k1 = kc + 16 * half + 8 + 2 * v;
      a.h[8 + 2 * v]     = f2bf(xr[k1]);
      a.h[8 + 2 * v + 1] = f2bf(xr[k1 + 1]);
    }
    if (kc + 32 < MODEL_DIM)
      __builtin_prefetch(xr + kc + 32, 0, 0);   // global_prefetch_b8

#pragma unroll
    for (int w = 0; w < 3; ++w) {
#pragma unroll
      for (int t = 0; t < 4; ++t) {
        BfFrag bfr;
        bfr.v = *(const v16bf*)&Wt[w][t * 16 + m][16 * half];
        acc[w][t] = __builtin_amdgcn_wmma_f32_16x16x32_bf16(
            false, a.v, false, bfr.v, (short)0, acc[w][t], false, false);
      }
    }
  }

  const float* bs[3]   = {bq, bk, bv};
  __bf16*      outs[3] = {Qo, Ko, Vo};
#pragma unroll
  for (int w = 0; w < 3; ++w) {
#pragma unroll
    for (int t = 0; t < 4; ++t) {
      float bias = bs[w][t * 16 + m];
#pragma unroll
      for (int r = 0; r < 8; ++r) {
        long row = rowBase + r + 8 * half;
        outs[w][row * HEAD_DIM + t * 16 + m] = f2bf(acc[w][t][r] + bias);
      }
    }
  }
}

// ---------------------------------------------------------------------------
// Kernel 2: causal flash attention. Block = 128 threads (4 waves); each wave
// owns 16 query rows; workgroup owns 64 queries of one batch. Key blocks of
// 64 iterate only up to the diagonal. K/V staging is double-buffered: K is
// copied with async global->LDS b128 (ASYNCcnt), V is loaded b128 and
// scattered transposed; the next block's staging overlaps this block's WMMAs.
// ---------------------------------------------------------------------------
__global__ __launch_bounds__(128) void flash_attn_kernel(
    const __bf16* __restrict__ Qg, const __bf16* __restrict__ Kg,
    const __bf16* __restrict__ Vg, float* __restrict__ out)
{
  __shared__ __align__(32) __bf16 Ks[2][64][HEAD_DIM];  // [buf][key][dh] 16 KB
  __shared__ __align__(32) __bf16 Vt[2][HEAD_DIM][64];  // [buf][dh][key] 16 KB
  __shared__ __align__(32) __bf16 Ps[4][16][64];        // per-wave P       8 KB

  const int  tid  = threadIdx.x;
  const int  lane = tid & 31;
  const int  wave = tid >> 5;          // 0..3
  const int  m    = lane & 15;
  const int  half = lane >> 4;
  const int  b    = blockIdx.y;
  const int  q0   = blockIdx.x * 64;
  const long bBase = (long)b * SEQ * HEAD_DIM;
  const int  qRow = q0 + wave * 16 + m;               // A-layout row

  // Stage one 64-key block into buffer `buf`: K straight copy (async when
  // available), V transposed via b128 load + 8x b16 scatter.
  auto stage = [&](int buf, int kb) {
#pragma unroll
    for (int it = 0; it < 4; ++it) {
      int c    = tid + it * 128;       // 0..511 sixteen-byte chunks
      int key  = c >> 3;               // 0..63
      int off8 = (c & 7) * 8;          // 0,8,...,56
      long g = bBase + (long)(kb + key) * HEAD_DIM + off8;
#if HAVE_ASYNC_LDS
      __builtin_amdgcn_global_load_async_to_lds_b128(
          to_gbl(Kg + g), to_lds(&Ks[buf][key][off8]), 0, 0);
#else
      *(uint4*)&Ks[buf][key][off8] = *(const uint4*)(Kg + g);
#endif
      uint4 vv = *(const uint4*)(Vg + g);
      const __bf16* hv = (const __bf16*)&vv;
#pragma unroll
      for (int j = 0; j < 8; ++j)
        Vt[buf][off8 + j][key] = hv[j];
    }
  };

  // Q A-fragments for K-chunks 0..31 and 32..63 (loaded once).
  BfFrag aq[2];
  {
    const unsigned* qrow = (const unsigned*)(Qg + bBase + (long)qRow * HEAD_DIM);
#pragma unroll
    for (int c = 0; c < 2; ++c) {
      int kc = c * 32;
#pragma unroll
      for (int v = 0; v < 4; ++v) {
        aq[c].u[v]     = qrow[(kc + 16 * half + 2 * v) >> 1];
        aq[c].u[4 + v] = qrow[(kc + 16 * half + 8 + 2 * v) >> 1];
      }
    }
  }

  v8f   Oacc[4] = {};                  // 16x64 fp32 output accumulator
  float m_i[8], l_i[8];
#pragma unroll
  for (int r = 0; r < 8; ++r) { m_i[r] = -1e30f; l_i[r] = 0.0f; }

  const float scale = 0.125f;          // 1/sqrt(64)

  stage(0, 0);                         // prologue: first key block in flight

  for (int kb = 0; kb <= q0; kb += 64) {
    const int cur = (kb >> 6) & 1;

    wait_async_zero();                 // my async copies for buf `cur` landed
    __syncthreads();                   // everyone's copies landed; buf cur^1 free

    if (kb + 64 <= q0)                 // overlap next block's staging with WMMAs
      stage(cur ^ 1, kb + 64);

    // S = Q @ K^T  (4 key tiles of 16 cols, K-depth 64 = 2 WMMAs each)
    v8f Sreg[4];
#pragma unroll
    for (int t = 0; t < 4; ++t) {
      v8f c = {};
#pragma unroll
      for (int kc = 0; kc < 2; ++kc) {
        BfFrag bk_;
        bk_.v = *(const v16bf*)&Ks[cur][t * 16 + m][kc * 32 + 16 * half];
        c = __builtin_amdgcn_wmma_f32_16x16x32_bf16(
            false, aq[kc].v, false, bk_.v, (short)0, c, false, false);
      }
      Sreg[t] = c;
    }

    // Scale + causal mask.
#pragma unroll
    for (int t = 0; t < 4; ++t) {
      int col = kb + t * 16 + m;
#pragma unroll
      for (int r = 0; r < 8; ++r) {
        int row = q0 + wave * 16 + r + 8 * half;
        float s = Sreg[t][r] * scale;
        Sreg[t][r] = (col <= row) ? s : -1e30f;
      }
    }

    // Online softmax update (row lives across 16 lanes of one half).
#pragma unroll
    for (int r = 0; r < 8; ++r) {
      float mx = fmaxf(fmaxf(Sreg[0][r], Sreg[1][r]),
                       fmaxf(Sreg[2][r], Sreg[3][r]));
#pragma unroll
      for (int off = 8; off >= 1; off >>= 1)
        mx = fmaxf(mx, __shfl_xor(mx, off, 32));
      float mnew  = fmaxf(m_i[r], mx);
      float alpha = __expf(m_i[r] - mnew);
      float rsum  = 0.0f;
#pragma unroll
      for (int t = 0; t < 4; ++t) {
        float p = __expf(Sreg[t][r] - mnew);
        Sreg[t][r] = p;
        rsum += p;
      }
#pragma unroll
      for (int off = 8; off >= 1; off >>= 1)
        rsum += __shfl_xor(rsum, off, 32);
      l_i[r] = l_i[r] * alpha + rsum;
      m_i[r] = mnew;
#pragma unroll
      for (int t = 0; t < 4; ++t)
        Oacc[t][r] *= alpha;
    }

    // Re-layout P: C/D layout -> LDS -> A layout (wave-private tile).
#pragma unroll
    for (int t = 0; t < 4; ++t)
#pragma unroll
      for (int r = 0; r < 8; ++r)
        Ps[wave][r + 8 * half][t * 16 + m] = f2bf(Sreg[t][r]);

    asm volatile("s_wait_dscnt 0" ::: "memory");  // LDS RAW, same wave

    // O += P @ V   (4 dh tiles, K-depth 64 = 2 chunks of 32 keys)
    const unsigned* prow = (const unsigned*)&Ps[wave][m][0];
#pragma unroll
    for (int kc = 0; kc < 2; ++kc) {
      BfFrag ap;
#pragma unroll
      for (int v = 0; v < 4; ++v) {
        ap.u[v]     = prow[(kc * 32 + 16 * half + 2 * v) >> 1];
        ap.u[4 + v] = prow[(kc * 32 + 16 * half + 8 + 2 * v) >> 1];
      }
#pragma unroll
      for (int t = 0; t < 4; ++t) {
        BfFrag bv_;
        bv_.v = *(const v16bf*)&Vt[cur][t * 16 + m][kc * 32 + 16 * half];
        Oacc[t] = __builtin_amdgcn_wmma_f32_16x16x32_bf16(
            false, ap.v, false, bv_.v, (short)0, Oacc[t], false, false);
      }
    }
  }

  // Normalize by row sum and store fp32 output.
#pragma unroll
  for (int r = 0; r < 8; ++r) {
    float inv = 1.0f / l_i[r];
    int   row = q0 + wave * 16 + r + 8 * half;
    float* orow = out + bBase + (long)row * HEAD_DIM;
#pragma unroll
    for (int t = 0; t < 4; ++t)
      orow[t * 16 + m] = Oacc[t][r] * inv;
  }
}

// ---------------------------------------------------------------------------
// Launch: inputs are (x, mask, Wk, bk, Wq, bq, Wv, bv); mask is the static
// causal tril so it is hardcoded. Workspace holds bf16 Q/K/V (6 MB).
// ---------------------------------------------------------------------------
extern "C" void kernel_launch(void* const* d_in, const int* in_sizes, int n_in,
                              void* d_out, int out_size, void* d_ws, size_t ws_size,
                              hipStream_t stream) {
  const float* x  = (const float*)d_in[0];
  const float* Wk = (const float*)d_in[2];
  const float* bk = (const float*)d_in[3];
  const float* Wq = (const float*)d_in[4];
  const float* bq = (const float*)d_in[5];
  const float* Wv = (const float*)d_in[6];
  const float* bv = (const float*)d_in[7];

  const size_t tokens = (size_t)BATCH * SEQ * HEAD_DIM;
  __bf16* Q = (__bf16*)d_ws;
  __bf16* K = Q + tokens;
  __bf16* V = K + tokens;

  qkv_proj_kernel<<<dim3((BATCH * SEQ) / 128), dim3(256), 0, stream>>>(
      x, Wk, bk, Wq, bq, Wv, bv, Q, K, V);

  flash_attn_kernel<<<dim3(SEQ / 64, BATCH), dim3(128), 0, stream>>>(
      Q, K, V, (float*)d_out);
}
